// PCLayer_47321949667586
// MI455X (gfx1250) — matrix operand order repaired and checked by reading, not compiled, not run on hardware
//
#include <hip/hip_runtime.h>

// ---------------------------------------------------------------------------
// PCLayer fixed-point inference on MI455X (gfx1250, wave32, WMMA).
//   T=10 steps of:  mu = x@W^T + b ; err = target - mu ; x = clip(x + lr*err@W)
// Mixed precision: GEMM operands bf16 (v_wmma_f32_16x16x32_bf16, f32 accum),
// state x / err epilogue math in f32.
// Mainloop uses gfx1250 async memory->LDS copies (ASYNCcnt) with a
// double-buffered software pipeline: one barrier per K-step.
// ---------------------------------------------------------------------------

typedef __attribute__((ext_vector_type(16))) __bf16 v16bf;
typedef __attribute__((ext_vector_type(8)))  __bf16 v8bf;
typedef __attribute__((ext_vector_type(8)))  float  v8f;

#define M_TOTAL 8192          // B*S = 8*1024 tokens
#define H_IN    768
#define H_OUT   3072
#define T_STEPS 10
#define LR      0.001f
#define CLAMPV  1.0f

#define BM 128                // block tile M
#define BN 128                // block tile N
#define BK 32                 // K step (bf16 WMMA K)
#define LDST 40               // padded LDS row stride (halves): 32 + 8 pad
#define TILE_HALVES (BM * LDST)
#define THREADS 256           // 8 waves of 32

// ---- gfx1250 async memory->LDS copy (VGLOBAL, tracked by ASYNCcnt) --------
__device__ __forceinline__ void async_copy_b128(unsigned lds_byte_addr,
                                                const void* gaddr) {
    asm volatile("global_load_async_to_lds_b128 %0, %1, off"
                 :: "v"(lds_byte_addr), "v"(gaddr)
                 : "memory");
}
__device__ __forceinline__ void async_wait0() {
    asm volatile("s_wait_asynccnt 0" ::: "memory");
}

// ---------------------------------------------------------------------------
// One-time prep: W (f32, [H_OUT][H_IN]) -> bf16 row-major Wb and bf16
// transpose WbT ([H_IN][H_OUT]). Folding the transpose here lets both GEMMs
// stage their B tiles as contiguous row-major copies.
// ---------------------------------------------------------------------------
__global__ void prep_w(const float* __restrict__ W,
                       __bf16* __restrict__ Wb,
                       __bf16* __restrict__ WbT) {
    const int total = H_OUT * H_IN;
    for (int idx = blockIdx.x * blockDim.x + threadIdx.x; idx < total;
         idx += gridDim.x * blockDim.x) {
        float w = W[idx];
        int r = idx / H_IN;
        int c = idx - r * H_IN;
        __bf16 wb = (__bf16)w;
        Wb[idx] = wb;
        WbT[(size_t)c * H_OUT + r] = wb;
    }
}

// x0 = 0 (both f32 master copy and bf16 mirror)
__global__ void zero_x(float* __restrict__ x, __bf16* __restrict__ xb) {
    const int total = M_TOTAL * H_IN;
    for (int idx = blockIdx.x * blockDim.x + threadIdx.x; idx < total;
         idx += gridDim.x * blockDim.x) {
        x[idx]  = 0.0f;
        xb[idx] = (__bf16)0.0f;
    }
}

// ---------------------------------------------------------------------------
// Double-buffered WMMA mainloop:
//   acc[4][2] += A[m0:m0+128, :K] * Bt[n0:n0+128, :K]^T
//   A  : row-major [*, lda]  (bf16)   -- A-operand rows
//   Bt : row-major [*, ldbt] (bf16)   -- column n of B stored as row n of Bt
// Each of the 8 waves computes a 64x32 sub-tile = 4x2 WMMA 16x16 tiles.
// Each thread owns 4 fixed 16B chunks of the (A,Bt) tiles; staging is four
// unconditional global_load_async_to_lds_b128 per K-step.
// ---------------------------------------------------------------------------
__device__ __forceinline__ void gemm_tile(const __bf16* __restrict__ A, int lda,
                                          const __bf16* __restrict__ Bt, int ldbt,
                                          int K, int m0, int n0,
                                          __bf16* As, __bf16* Bs,
                                          v8f (&acc)[4][2]) {
    const int tid   = threadIdx.x;
    const int lane  = tid & 31;
    const int w     = tid >> 5;     // wave 0..7
    const int wm    = w & 1;        // M half (64 rows each)
    const int wn    = w >> 1;       // N quarter (32 cols each)
    const int laneM = lane & 15;
    const int hi    = lane >> 4;    // lane group 0/1

    // ---- this thread's staging chunks: rows row0/row0+64, 8-half segment ----
    const int row0 = tid >> 2;              // 0..63
    const int row1 = row0 + 64;             // 64..127
    const int seg  = (tid & 3) * 8;         // half offset within row: 0/8/16/24
    const __bf16* Ap0 = A  + (size_t)(m0 + row0) * lda  + seg;
    const __bf16* Ap1 = A  + (size_t)(m0 + row1) * lda  + seg;
    const __bf16* Bp0 = Bt + (size_t)(n0 + row0) * ldbt + seg;
    const __bf16* Bp1 = Bt + (size_t)(n0 + row1) * ldbt + seg;
    // LDS byte addresses (flat LDS aperture keeps LDS offset in addr[31:0])
    const unsigned aAs = (unsigned)(size_t)As;
    const unsigned aBs = (unsigned)(size_t)Bs;
    const unsigned lA0 = aAs + 2u * (unsigned)(row0 * LDST + seg);
    const unsigned lA1 = aAs + 2u * (unsigned)(row1 * LDST + seg);
    const unsigned lB0 = aBs + 2u * (unsigned)(row0 * LDST + seg);
    const unsigned lB1 = aBs + 2u * (unsigned)(row1 * LDST + seg);

    auto stage = [&](int kt, int b) {
        const unsigned boff = (unsigned)(b * TILE_HALVES * 2);
        async_copy_b128(lA0 + boff, Ap0 + kt);
        async_copy_b128(lA1 + boff, Ap1 + kt);
        async_copy_b128(lB0 + boff, Bp0 + kt);
        async_copy_b128(lB1 + boff, Bp1 + kt);
    };

    stage(0, 0);                 // prologue prefetch
    int b = 0;
    for (int kt = 0; kt < K; kt += BK) {
        async_wait0();           // this wave's async copies done
        __syncthreads();         // -> entire tile resident
        if (kt + BK < K) stage(kt + BK, b ^ 1);

        const __bf16* Ab = As + b * TILE_HALVES;
        const __bf16* Bb = Bs + b * TILE_HALVES;

        // ---- A frags: lanes 0-15 hold K 0-7 & 16-23; lanes 16-31 K 8-15 & 24-31
        v16bf afr[4];
        #pragma unroll
        for (int tm = 0; tm < 4; ++tm) {
            int row = wm * 64 + tm * 16 + laneM;
            v8bf lo  = *(const v8bf*)&Ab[row * LDST + hi * 8];
            v8bf hi8 = *(const v8bf*)&Ab[row * LDST + 16 + hi * 8];
            afr[tm] = __builtin_shufflevector(lo, hi8, 0, 1, 2, 3, 4, 5, 6, 7,
                                              8, 9, 10, 11, 12, 13, 14, 15);
        }
        // ---- B frags: lanes 0-15 hold K 0-7 & 16-23; lanes 16-31 K 8-15 & 24-31 (per column)
        v16bf bfr[2];
        #pragma unroll
        for (int tn = 0; tn < 2; ++tn) {
            int col = wn * 32 + tn * 16 + laneM;
            v8bf lo  = *(const v8bf*)&Bb[col * LDST + hi * 8];
            v8bf hi8 = *(const v8bf*)&Bb[col * LDST + 16 + hi * 8];
            bfr[tn] = __builtin_shufflevector(lo, hi8, 0, 1, 2, 3, 4, 5, 6, 7,
                                              8, 9, 10, 11, 12, 13, 14, 15);
        }
        // ---- 8 WMMAs: D = A*B + C (f32 accumulate) ----
        #pragma unroll
        for (int tm = 0; tm < 4; ++tm)
            #pragma unroll
            for (int tn = 0; tn < 2; ++tn)
                acc[tm][tn] = __builtin_amdgcn_wmma_f32_16x16x32_bf16(
                    false, afr[tm], false, bfr[tn], (short)0, acc[tm][tn],
                    false, false);
        b ^= 1;
    }
}

// ---------------------------------------------------------------------------
// Step kernel 1: err = target - (x @ W^T + b), stored bf16 (A-operand of k_upd)
// A = xb [8192][768], Bt = Wb [3072][768] (row n of Wb == column n of W^T)
// ---------------------------------------------------------------------------
__global__ void k_err(const __bf16* __restrict__ xb,
                      const __bf16* __restrict__ Wb,
                      const float* __restrict__ target,
                      const float* __restrict__ bvec,
                      __bf16* __restrict__ errb) {
    __shared__ __align__(16) __bf16 As[2 * TILE_HALVES];
    __shared__ __align__(16) __bf16 Bs[2 * TILE_HALVES];

    const int m0 = blockIdx.x * BM;
    const int n0 = blockIdx.y * BN;

    v8f acc[4][2];
    #pragma unroll
    for (int i = 0; i < 4; ++i)
        #pragma unroll
        for (int j = 0; j < 2; ++j) acc[i][j] = (v8f)0.0f;

    gemm_tile(xb, H_IN, Wb, H_IN, H_IN, m0, n0, As, Bs, acc);

    const int lane  = threadIdx.x & 31;
    const int w     = threadIdx.x >> 5;
    const int wm    = w & 1;
    const int wn    = w >> 1;
    const int laneM = lane & 15;
    const int hi    = lane >> 4;

    #pragma unroll
    for (int tm = 0; tm < 4; ++tm) {
        const int rowb = m0 + wm * 64 + tm * 16 + hi * 8;
        #pragma unroll
        for (int tn = 0; tn < 2; ++tn) {
            const int col = n0 + wn * 32 + tn * 16 + laneM;
            const float bc = bvec[col];
            #pragma unroll
            for (int r = 0; r < 8; ++r) {
                size_t off = (size_t)(rowb + r) * H_OUT + col;
                float e = target[off] - (acc[tm][tn][r] + bc);
                errb[off] = (__bf16)e;
            }
        }
    }
}

// ---------------------------------------------------------------------------
// Step kernel 2: x = clip(x + lr * err @ W); refresh bf16 mirror xb
// A = errb [8192][3072], Bt = WbT [768][3072] (row n of WbT == column n of W)
// ---------------------------------------------------------------------------
__global__ void k_upd(const __bf16* __restrict__ errb,
                      const __bf16* __restrict__ WbT,
                      float* __restrict__ x,
                      __bf16* __restrict__ xb) {
    __shared__ __align__(16) __bf16 As[2 * TILE_HALVES];
    __shared__ __align__(16) __bf16 Bs[2 * TILE_HALVES];

    const int m0 = blockIdx.x * BM;
    const int n0 = blockIdx.y * BN;

    v8f acc[4][2];
    #pragma unroll
    for (int i = 0; i < 4; ++i)
        #pragma unroll
        for (int j = 0; j < 2; ++j) acc[i][j] = (v8f)0.0f;

    gemm_tile(errb, H_OUT, WbT, H_OUT, H_OUT, m0, n0, As, Bs, acc);

    const int lane  = threadIdx.x & 31;
    const int w     = threadIdx.x >> 5;
    const int wm    = w & 1;
    const int wn    = w >> 1;
    const int laneM = lane & 15;
    const int hi    = lane >> 4;

    #pragma unroll
    for (int tm = 0; tm < 4; ++tm) {
        const int rowb = m0 + wm * 64 + tm * 16 + hi * 8;
        #pragma unroll
        for (int tn = 0; tn < 2; ++tn) {
            const int col = n0 + wn * 32 + tn * 16 + laneM;
            #pragma unroll
            for (int r = 0; r < 8; ++r) {
                size_t off = (size_t)(rowb + r) * H_IN + col;
                float xv = x[off] + LR * acc[tm][tn][r];
                xv = fminf(CLAMPV, fmaxf(-CLAMPV, xv));
                x[off]  = xv;
                xb[off] = (__bf16)xv;
            }
        }
    }
}

// ---------------------------------------------------------------------------
// Host launcher. Inputs: [0]=target_activity f32 [8,1024,3072],
// [1]=W f32 [3072,768], [2]=b f32 [3072]. Output: x f32 [8,1024,768].
// d_out doubles as the persistent f32 x state across the 10 steps.
// ---------------------------------------------------------------------------
extern "C" void kernel_launch(void* const* d_in, const int* in_sizes, int n_in,
                              void* d_out, int out_size, void* d_ws, size_t ws_size,
                              hipStream_t stream) {
    const float* target = (const float*)d_in[0];
    const float* W      = (const float*)d_in[1];
    const float* bvec   = (const float*)d_in[2];
    float* x = (float*)d_out;

    char* ws = (char*)d_ws;
    const size_t SZ_W  = (size_t)H_OUT * H_IN * sizeof(__bf16);   // 4.5 MiB
    const size_t SZ_XB = (size_t)M_TOTAL * H_IN * sizeof(__bf16); // 12 MiB
    __bf16* Wb   = (__bf16*)(ws);
    __bf16* WbT  = (__bf16*)(ws + SZ_W);
    __bf16* xb   = (__bf16*)(ws + 2 * SZ_W);
    __bf16* errb = (__bf16*)(ws + 2 * SZ_W + SZ_XB);              // 48 MiB

    prep_w<<<1024, 256, 0, stream>>>(W, Wb, WbT);
    zero_x<<<4096, 256, 0, stream>>>(x, xb);

    dim3 g_err(M_TOTAL / BM, H_OUT / BN);  // 64 x 24
    dim3 g_upd(M_TOTAL / BM, H_IN / BN);   // 64 x 6
    for (int t = 0; t < T_STEPS; ++t) {
        k_err<<<g_err, THREADS, 0, stream>>>(xb, Wb, target, bvec, errb);
        k_upd<<<g_upd, THREADS, 0, stream>>>(errb, WbT, x, xb);
    }
}
